// EncodeProcessDecode_31215822308103
// MI455X (gfx1250) — compile-verified, matrix operand order checked by reading, not run on hardware
//
#include <hip/hip_runtime.h>

typedef __attribute__((ext_vector_type(16))) _Float16 v16h;
typedef __attribute__((ext_vector_type(8)))  _Float16 v8h;
typedef __attribute__((ext_vector_type(8)))  float    v8f;

union V16 { v16h v; v8h h[2]; };

#define DEVINL __device__ __forceinline__
#define MLP_WAVES 8

DEVINL v8f v8f_zero() {
  v8f z;
#pragma unroll
  for (int i = 0; i < 8; ++i) z[i] = 0.f;
  return z;
}

DEVINL v8h v8h_zero() {
  v8h z;
#pragma unroll
  for (int i = 0; i < 8; ++i) z[i] = (_Float16)0.f;
  return z;
}

DEVINL v8f wmma32(v16h a, v16h b, v8f c) {
  // D = A(16x32 f16) x B(32x16 f16) + C(16x16 f32)
  return __builtin_amdgcn_wmma_f32_16x16x32_f16(false, a, false, b, (short)0, c,
                                                false, false);
}

// B fragment: packed blocks of (32 lanes x 16 halves), block index = kc*nT + nt
DEVINL v16h load_b_frag(const _Float16* wp, int lane, int blk) {
  return *(const v16h*)(wp + ((size_t)blk * 32 + lane) * 16);
}

// A fragment from 16 consecutive rows of 64 halves (LDS tile or global table)
DEVINL v16h load_a_row64(const _Float16* base, int lane, int kc) {
  const int row = lane & 15;
  const int b0  = (lane >> 4) << 3;   // 0 for lanes 0-15, 8 for lanes 16-31
  const _Float16* p = base + (size_t)row * 64 + kc * 32;
  V16 u;
  u.h[0] = *(const v8h*)(p + b0);
  u.h[1] = *(const v8h*)(p + 16 + b0);
  return u.v;
}

DEVINL void store_act(_Float16* A, v8f c[4], const float* bias, int lane, bool relu) {
  const int col0 = lane & 15;
  const int rb   = (lane >> 4) << 3;  // D layout: lanes>=16 hold rows 8..15
#pragma unroll
  for (int nt = 0; nt < 4; ++nt) {
    const float bv = bias[col0 + nt * 16];
#pragma unroll
    for (int r = 0; r < 8; ++r) {
      float x = c[nt][r] + bv;
      if (relu) x = x > 0.f ? x : 0.01f * x;
      A[(rb + r) * 64 + col0 + nt * 16] = (_Float16)x;
    }
  }
}

DEVINL void hidden64(const _Float16* Ain, _Float16* Aout, const _Float16* wp,
                     const float* bias, int lane) {
  v8f c[4];
#pragma unroll
  for (int nt = 0; nt < 4; ++nt) c[nt] = v8f_zero();
#pragma unroll
  for (int kc = 0; kc < 2; ++kc) {
    v16h a = load_a_row64(Ain, lane, kc);
#pragma unroll
    for (int nt = 0; nt < 4; ++nt)
      c[nt] = wmma32(a, load_b_frag(wp, lane, kc * 4 + nt), c[nt]);
  }
  store_act(Aout, c, bias, lane, true);
}

// last linear layer (64->64) + layernorm; result left in c[4]
DEVINL void final_ln(const _Float16* Ain, const _Float16* wp, const float* bias,
                     const float* gamma, const float* beta, int lane, v8f c[4]) {
#pragma unroll
  for (int nt = 0; nt < 4; ++nt) c[nt] = v8f_zero();
#pragma unroll
  for (int kc = 0; kc < 2; ++kc) {
    v16h a = load_a_row64(Ain, lane, kc);
#pragma unroll
    for (int nt = 0; nt < 4; ++nt)
      c[nt] = wmma32(a, load_b_frag(wp, lane, kc * 4 + nt), c[nt]);
  }
  const int col0 = lane & 15;
  float bv[4], gv[4], be[4];
#pragma unroll
  for (int nt = 0; nt < 4; ++nt) {
    bv[nt] = bias[col0 + nt * 16];
    gv[nt] = gamma[col0 + nt * 16];
    be[nt] = beta[col0 + nt * 16];
  }
#pragma unroll
  for (int r = 0; r < 8; ++r) {
    float x0 = c[0][r] + bv[0];
    float x1 = c[1][r] + bv[1];
    float x2 = c[2][r] + bv[2];
    float x3 = c[3][r] + bv[3];
    float s = x0 + x1 + x2 + x3;
    float q = x0 * x0 + x1 * x1 + x2 * x2 + x3 * x3;
    s += __shfl_xor(s, 1, 16); s += __shfl_xor(s, 2, 16);
    s += __shfl_xor(s, 4, 16); s += __shfl_xor(s, 8, 16);
    q += __shfl_xor(q, 1, 16); q += __shfl_xor(q, 2, 16);
    q += __shfl_xor(q, 4, 16); q += __shfl_xor(q, 8, 16);
    const float mu  = s * (1.f / 64.f);
    const float var = q * (1.f / 64.f) - mu * mu;
    const float rs  = rsqrtf(var + 1e-5f);
    c[0][r] = (x0 - mu) * rs * gv[0] + be[0];
    c[1][r] = (x1 - mu) * rs * gv[1] + be[1];
    c[2][r] = (x2 - mu) * rs * gv[2] + be[2];
    c[3][r] = (x3 - mu) * rs * gv[3] + be[3];
  }
}

struct ProcW {
  const _Float16 *w0, *w1, *w2, *w3;
  const float *b0, *b1, *b2, *b3, *gamma, *beta;
};

// ---------------- layer-1 gather helpers ----------------

// edge_enc input: concat(edges[8], nodes[s][16], nodes[r][16]), Kpad=64
DEVINL v8h ee_sub(int o8, const _Float16* edgesH, const _Float16* nodesH,
                  int s, int rv, int e) {
  switch (o8) {
    case 0: return *(const v8h*)(edgesH + (size_t)e * 8);
    case 1: return *(const v8h*)(nodesH + (size_t)s * 16);
    case 2: return *(const v8h*)(nodesH + (size_t)s * 16 + 8);
    case 3: return *(const v8h*)(nodesH + (size_t)rv * 16);
    case 4: return *(const v8h*)(nodesH + (size_t)rv * 16 + 8);
    default: return v8h_zero();
  }
}

// node_enc input: concat(nodes[i][16], agg0[i][64]), Kpad=96
DEVINL v8h ne_sub(int o8, const _Float16* nodesH, const _Float16* agg0H, int i) {
  if (o8 >= 10) return v8h_zero();
  if (o8 == 0) return *(const v8h*)(nodesH + (size_t)i * 16);
  if (o8 == 1) return *(const v8h*)(nodesH + (size_t)i * 16 + 8);
  return *(const v8h*)(agg0H + (size_t)i * 64 + (size_t)(o8 - 2) * 8);
}

// edge_proc input: concat(latN[s], latN[r], latE[e], encN[s], encN[r], encE[e]), K=384
DEVINL v8h ep_sub(int o8, const _Float16* latN, const _Float16* latE,
                  const _Float16* encN, const _Float16* encE, int s, int rv, int e) {
  const int off = (o8 & 7) * 8;
  const _Float16* p;
  switch (o8 >> 3) {
    case 0:  p = latN + (size_t)s  * 64; break;
    case 1:  p = latN + (size_t)rv * 64; break;
    case 2:  p = latE + (size_t)e  * 64; break;
    case 3:  p = encN + (size_t)s  * 64; break;
    case 4:  p = encN + (size_t)rv * 64; break;
    default: p = encE + (size_t)e  * 64; break;
  }
  return *(const v8h*)(p + off);
}

// node_proc input: concat(latN[i], agg[i], encN[i], agg0[i]), K=256
DEVINL v8h np_sub(int o8, const _Float16* latN, const _Float16* aggH,
                  const _Float16* encN, const _Float16* agg0H, int i) {
  const int off = (o8 & 7) * 8;
  const _Float16* p;
  switch (o8 >> 3) {
    case 0:  p = latN  + (size_t)i * 64; break;
    case 1:  p = aggH  + (size_t)i * 64; break;
    case 2:  p = encN  + (size_t)i * 64; break;
    default: p = agg0H + (size_t)i * 64; break;
  }
  return *(const v8h*)(p + off);
}

// ---------------- MLP kernels ----------------

__global__ __launch_bounds__(256) void k_edge_enc(
    const _Float16* __restrict__ nodesH, const _Float16* __restrict__ edgesH,
    const int* __restrict__ senders, const int* __restrict__ receivers,
    ProcW w, _Float16* __restrict__ encE, float* aggF, int E) {
  __shared__ _Float16 act[MLP_WAVES][16 * 64];
  const int lane = threadIdx.x & 31;
  const int wv   = threadIdx.x >> 5;
  const int tile = blockIdx.x * MLP_WAVES + wv;
  if (tile * 16 >= E) return;
  _Float16* A = act[wv];
  const int row = lane & 15;
  const int b0  = (lane >> 4) << 3;
  const int e   = tile * 16 + row;
  const int s   = senders[e];
  const int rv  = receivers[e];

  v8f c[4];
#pragma unroll
  for (int nt = 0; nt < 4; ++nt) c[nt] = v8f_zero();
#pragma unroll
  for (int kc = 0; kc < 2; ++kc) {
    V16 u;
    const int o8 = kc * 4 + (b0 >> 3);
    u.h[0] = ee_sub(o8,     edgesH, nodesH, s, rv, e);
    u.h[1] = ee_sub(o8 + 2, edgesH, nodesH, s, rv, e);
#pragma unroll
    for (int nt = 0; nt < 4; ++nt)
      c[nt] = wmma32(u.v, load_b_frag(w.w0, lane, kc * 4 + nt), c[nt]);
  }
  store_act(A, c, w.b0, lane, true);
  hidden64(A, A, w.w1, w.b1, lane);
  hidden64(A, A, w.w2, w.b2, lane);
  final_ln(A, w.w3, w.b3, w.gamma, w.beta, lane, c);

  const int col0 = lane & 15;
  int ridx[8];
#pragma unroll
  for (int r = 0; r < 8; ++r) ridx[r] = receivers[tile * 16 + b0 + r];
#pragma unroll
  for (int nt = 0; nt < 4; ++nt) {
    const int col = col0 + nt * 16;
#pragma unroll
    for (int r = 0; r < 8; ++r) {
      const float v = c[nt][r];
      encE[(size_t)(tile * 16 + b0 + r) * 64 + col] = (_Float16)v;
      unsafeAtomicAdd(&aggF[(size_t)ridx[r] * 64 + col], v);
    }
  }
}

__global__ __launch_bounds__(256) void k_node_enc(
    const _Float16* __restrict__ nodesH, const _Float16* __restrict__ agg0H,
    ProcW w, _Float16* __restrict__ encN, int N) {
  __shared__ _Float16 act[MLP_WAVES][16 * 64];
  const int lane = threadIdx.x & 31;
  const int wv   = threadIdx.x >> 5;
  const int tile = blockIdx.x * MLP_WAVES + wv;
  if (tile * 16 >= N) return;
  _Float16* A = act[wv];
  const int row = lane & 15;
  const int b0  = (lane >> 4) << 3;
  const int i   = tile * 16 + row;

  v8f c[4];
#pragma unroll
  for (int nt = 0; nt < 4; ++nt) c[nt] = v8f_zero();
#pragma unroll
  for (int kc = 0; kc < 3; ++kc) {
    V16 u;
    const int o8 = kc * 4 + (b0 >> 3);
    u.h[0] = ne_sub(o8,     nodesH, agg0H, i);
    u.h[1] = ne_sub(o8 + 2, nodesH, agg0H, i);
#pragma unroll
    for (int nt = 0; nt < 4; ++nt)
      c[nt] = wmma32(u.v, load_b_frag(w.w0, lane, kc * 4 + nt), c[nt]);
  }
  store_act(A, c, w.b0, lane, true);
  hidden64(A, A, w.w1, w.b1, lane);
  hidden64(A, A, w.w2, w.b2, lane);
  final_ln(A, w.w3, w.b3, w.gamma, w.beta, lane, c);

  const int col0 = lane & 15;
#pragma unroll
  for (int nt = 0; nt < 4; ++nt) {
    const int col = col0 + nt * 16;
#pragma unroll
    for (int r = 0; r < 8; ++r)
      encN[(size_t)(tile * 16 + b0 + r) * 64 + col] = (_Float16)c[nt][r];
  }
}

__global__ __launch_bounds__(256) void k_edge_proc(
    const _Float16* latNin, const _Float16* latEin,
    const _Float16* __restrict__ encN, const _Float16* __restrict__ encE,
    const int* __restrict__ senders, const int* __restrict__ receivers,
    ProcW w, _Float16* latEout, float* aggF, int E) {
  __shared__ _Float16 act[MLP_WAVES][16 * 64];
  const int lane = threadIdx.x & 31;
  const int wv   = threadIdx.x >> 5;
  const int tile = blockIdx.x * MLP_WAVES + wv;
  if (tile * 16 >= E) return;
  _Float16* A = act[wv];
  const int row = lane & 15;
  const int b0  = (lane >> 4) << 3;
  const int e   = tile * 16 + row;
  const int s   = senders[e];
  const int rv  = receivers[e];

  v8f c[4];
#pragma unroll
  for (int nt = 0; nt < 4; ++nt) c[nt] = v8f_zero();
#pragma unroll
  for (int kc = 0; kc < 12; ++kc) {
    V16 u;
    const int o8 = kc * 4 + (b0 >> 3);
    u.h[0] = ep_sub(o8,     latNin, latEin, encN, encE, s, rv, e);
    u.h[1] = ep_sub(o8 + 2, latNin, latEin, encN, encE, s, rv, e);
#pragma unroll
    for (int nt = 0; nt < 4; ++nt)
      c[nt] = wmma32(u.v, load_b_frag(w.w0, lane, kc * 4 + nt), c[nt]);
  }
  store_act(A, c, w.b0, lane, true);
  hidden64(A, A, w.w1, w.b1, lane);
  hidden64(A, A, w.w2, w.b2, lane);
  final_ln(A, w.w3, w.b3, w.gamma, w.beta, lane, c);

  const int col0 = lane & 15;
  int ridx[8];
#pragma unroll
  for (int r = 0; r < 8; ++r) ridx[r] = receivers[tile * 16 + b0 + r];
#pragma unroll
  for (int nt = 0; nt < 4; ++nt) {
    const int col = col0 + nt * 16;
#pragma unroll
    for (int r = 0; r < 8; ++r) {
      const float v = c[nt][r];
      latEout[(size_t)(tile * 16 + b0 + r) * 64 + col] = (_Float16)v;
      unsafeAtomicAdd(&aggF[(size_t)ridx[r] * 64 + col], v);
    }
  }
}

__global__ __launch_bounds__(256) void k_node_proc(
    const _Float16* latNin, const _Float16* __restrict__ aggH,
    const _Float16* __restrict__ encN, const _Float16* __restrict__ agg0H,
    ProcW w, _Float16* latNout, int N) {
  __shared__ _Float16 act[MLP_WAVES][16 * 64];
  const int lane = threadIdx.x & 31;
  const int wv   = threadIdx.x >> 5;
  const int tile = blockIdx.x * MLP_WAVES + wv;
  if (tile * 16 >= N) return;
  _Float16* A = act[wv];
  const int row = lane & 15;
  const int b0  = (lane >> 4) << 3;
  const int i   = tile * 16 + row;

  v8f c[4];
#pragma unroll
  for (int nt = 0; nt < 4; ++nt) c[nt] = v8f_zero();
#pragma unroll
  for (int kc = 0; kc < 8; ++kc) {
    V16 u;
    const int o8 = kc * 4 + (b0 >> 3);
    u.h[0] = np_sub(o8,     latNin, aggH, encN, agg0H, i);
    u.h[1] = np_sub(o8 + 2, latNin, aggH, encN, agg0H, i);
#pragma unroll
    for (int nt = 0; nt < 4; ++nt)
      c[nt] = wmma32(u.v, load_b_frag(w.w0, lane, kc * 4 + nt), c[nt]);
  }
  store_act(A, c, w.b0, lane, true);
  hidden64(A, A, w.w1, w.b1, lane);
  hidden64(A, A, w.w2, w.b2, lane);
  final_ln(A, w.w3, w.b3, w.gamma, w.beta, lane, c);

  const int col0 = lane & 15;
#pragma unroll
  for (int nt = 0; nt < 4; ++nt) {
    const int col = col0 + nt * 16;
#pragma unroll
    for (int r = 0; r < 8; ++r)
      latNout[(size_t)(tile * 16 + b0 + r) * 64 + col] = (_Float16)c[nt][r];
  }
}

__global__ __launch_bounds__(256) void k_node_dec(
    const _Float16* __restrict__ latN, ProcW w, float* __restrict__ out, int N) {
  __shared__ _Float16 act[MLP_WAVES][16 * 64];
  const int lane = threadIdx.x & 31;
  const int wv   = threadIdx.x >> 5;
  const int tile = blockIdx.x * MLP_WAVES + wv;
  if (tile * 16 >= N) return;
  _Float16* A = act[wv];
  const int b0 = (lane >> 4) << 3;

  // layer 1: A straight from the latent table (16 consecutive rows of 64)
  const _Float16* base = latN + (size_t)tile * 16 * 64;
  v8f c[4];
#pragma unroll
  for (int nt = 0; nt < 4; ++nt) c[nt] = v8f_zero();
#pragma unroll
  for (int kc = 0; kc < 2; ++kc) {
    v16h a = load_a_row64(base, lane, kc);
#pragma unroll
    for (int nt = 0; nt < 4; ++nt)
      c[nt] = wmma32(a, load_b_frag(w.w0, lane, kc * 4 + nt), c[nt]);
  }
  store_act(A, c, w.b0, lane, true);
  hidden64(A, A, w.w1, w.b1, lane);
  hidden64(A, A, w.w2, w.b2, lane);

  // final 64 -> 3 (padded to 16 cols), linear
  v8f cf = v8f_zero();
#pragma unroll
  for (int kc = 0; kc < 2; ++kc) {
    v16h a = load_a_row64(A, lane, kc);
    cf = wmma32(a, load_b_frag(w.w3, lane, kc), cf);
  }
  const int col0 = lane & 15;
  if (col0 < 3) {
    const float bv = w.b3[col0];
#pragma unroll
    for (int r = 0; r < 8; ++r)
      out[(size_t)(tile * 16 + b0 + r) * 3 + col0] = cf[r] + bv;
  }
}

// ---------------- utility kernels ----------------

__global__ __launch_bounds__(256) void k_f32_to_f16(const float* __restrict__ s,
                                                    _Float16* __restrict__ d, int n) {
  int i = blockIdx.x * 256 + threadIdx.x;
  if (i < n) d[i] = (_Float16)s[i];
}

__global__ __launch_bounds__(256) void k_zero_f32(float* p, int n) {
  int i = blockIdx.x * 256 + threadIdx.x;
  if (i < n) p[i] = 0.f;
}

__global__ __launch_bounds__(256) void k_cvt_zero(float* s, _Float16* d, int n) {
  int i = blockIdx.x * 256 + threadIdx.x;
  if (i < n) {
    d[i] = (_Float16)s[i];
    s[i] = 0.f;
  }
}

// pack W[Kact x Nact] (row-major, K=rows) into WMMA-B fragment blocks
__global__ __launch_bounds__(256) void k_pack_w(const float* __restrict__ W,
                                                _Float16* __restrict__ dst,
                                                int Kact, int Kpad, int Nact, int Npad) {
  const int tid = blockIdx.x * 256 + threadIdx.x;
  const int total = (Kpad / 32) * (Npad / 16) * 512;
  if (tid >= total) return;
  const int h    = tid & 15;
  const int lane = (tid >> 4) & 31;
  const int blk  = tid >> 9;
  const int nT   = Npad / 16;
  const int kc   = blk / nT;
  const int nt   = blk % nT;
  const int b0   = (lane >> 4) << 3;
  const int k = kc * 32 + ((h < 8) ? (b0 + h) : (16 + b0 + (h - 8)));
  const int n = nt * 16 + (lane & 15);
  const float v = (k < Kact && n < Nact) ? W[(size_t)k * Nact + n] : 0.f;
  dst[tid] = (_Float16)v;
}

// ---------------- host ----------------

enum {
  EE_G = 2, EE_B = 3, EE_W0 = 4, EE_b0 = 5, EE_W1 = 6, EE_b1 = 7,
  EE_W2 = 8, EE_b2 = 9, EE_W3 = 10, EE_b3 = 11,
  EP_G = 12, EP_B = 13, EP_W0 = 14, EP_b0 = 15, EP_W1 = 16, EP_b1 = 17,
  EP_W2 = 18, EP_b2 = 19, EP_W3 = 20, EP_b3 = 21,
  ND_W0 = 22, ND_b0 = 23, ND_W1 = 24, ND_b1 = 25, ND_W2 = 26, ND_b2 = 27,
  ND_W3 = 28, ND_b3 = 29,
  NE_G = 30, NE_B = 31, NE_W0 = 32, NE_b0 = 33, NE_W1 = 34, NE_b1 = 35,
  NE_W2 = 36, NE_b2 = 37, NE_W3 = 38, NE_b3 = 39,
  NP_G = 40, NP_B = 41, NP_W0 = 42, NP_b0 = 43, NP_W1 = 44, NP_b1 = 45,
  NP_W2 = 46, NP_b2 = 47, NP_W3 = 48, NP_b3 = 49,
  IN_SND = 50, IN_RCV = 51
};

extern "C" void kernel_launch(void* const* d_in, const int* in_sizes, int n_in,
                              void* d_out, int out_size, void* d_ws, size_t ws_size,
                              hipStream_t stream) {
  (void)n_in; (void)out_size; (void)ws_size;
  const float* nodes = (const float*)d_in[0];
  const float* edges = (const float*)d_in[1];
  const int* senders   = (const int*)d_in[IN_SND];
  const int* receivers = (const int*)d_in[IN_RCV];
  const int N = in_sizes[0] / 16;
  const int E = in_sizes[1] / 8;
  const int STEPS = 3;
  float* out = (float*)d_out;

  auto Pf = [&](int i) { return (const float*)d_in[i]; };

  char* ws = (char*)d_ws;
  size_t off = 0;
  auto alloc = [&](size_t bytes) -> void* {
    off = (off + 255) & ~(size_t)255;
    void* p = ws + off;
    off += bytes;
    return p;
  };

  _Float16* nodesH = (_Float16*)alloc((size_t)N * 16 * 2);
  _Float16* edgesH = (_Float16*)alloc((size_t)E * 8 * 2);
  _Float16* encE   = (_Float16*)alloc((size_t)E * 64 * 2);
  _Float16* encN   = (_Float16*)alloc((size_t)N * 64 * 2);
  _Float16* latE   = (_Float16*)alloc((size_t)E * 64 * 2);
  _Float16* latN   = (_Float16*)alloc((size_t)N * 64 * 2);
  float*    agg0F  = (float*)alloc((size_t)N * 64 * 4);
  _Float16* agg0H  = (_Float16*)alloc((size_t)N * 64 * 2);
  float*    aggF   = (float*)alloc((size_t)N * 64 * 4);
  _Float16* aggH   = (_Float16*)alloc((size_t)N * 64 * 2);
  _Float16* wpack  = (_Float16*)alloc((size_t)131072 * 2);

  size_t wo = 0;
  auto packW = [&](int leaf, int Kact, int Kpad, int Nact, int Npad) -> _Float16* {
    _Float16* dst = wpack + wo;
    const int totalHalves = (Kpad / 32) * (Npad / 16) * 512;
    wo += (size_t)totalHalves;
    k_pack_w<<<dim3((totalHalves + 255) / 256), dim3(256), 0, stream>>>(
        (const float*)d_in[leaf], dst, Kact, Kpad, Nact, Npad);
    return dst;
  };

  // input conversion + aggregation buffer init
  k_f32_to_f16<<<dim3((N * 16 + 255) / 256), dim3(256), 0, stream>>>(nodes, nodesH, N * 16);
  k_f32_to_f16<<<dim3((E * 8 + 255) / 256), dim3(256), 0, stream>>>(edges, edgesH, E * 8);
  k_zero_f32<<<dim3((N * 64 + 255) / 256), dim3(256), 0, stream>>>(agg0F, N * 64);
  k_zero_f32<<<dim3((N * 64 + 255) / 256), dim3(256), 0, stream>>>(aggF, N * 64);

  // weight packing (K padded to mult. of 32, N to mult. of 16, zeros elsewhere)
  _Float16* eeW0 = packW(EE_W0, 40, 64, 64, 64);
  _Float16* eeW1 = packW(EE_W1, 64, 64, 64, 64);
  _Float16* eeW2 = packW(EE_W2, 64, 64, 64, 64);
  _Float16* eeW3 = packW(EE_W3, 64, 64, 64, 64);
  _Float16* neW0 = packW(NE_W0, 80, 96, 64, 64);
  _Float16* neW1 = packW(NE_W1, 64, 64, 64, 64);
  _Float16* neW2 = packW(NE_W2, 64, 64, 64, 64);
  _Float16* neW3 = packW(NE_W3, 64, 64, 64, 64);
  _Float16* epW0 = packW(EP_W0, 384, 384, 64, 64);
  _Float16* epW1 = packW(EP_W1, 64, 64, 64, 64);
  _Float16* epW2 = packW(EP_W2, 64, 64, 64, 64);
  _Float16* epW3 = packW(EP_W3, 64, 64, 64, 64);
  _Float16* npW0 = packW(NP_W0, 256, 256, 64, 64);
  _Float16* npW1 = packW(NP_W1, 64, 64, 64, 64);
  _Float16* npW2 = packW(NP_W2, 64, 64, 64, 64);
  _Float16* npW3 = packW(NP_W3, 64, 64, 64, 64);
  _Float16* ndW0 = packW(ND_W0, 64, 64, 64, 64);
  _Float16* ndW1 = packW(ND_W1, 64, 64, 64, 64);
  _Float16* ndW2 = packW(ND_W2, 64, 64, 64, 64);
  _Float16* ndW3 = packW(ND_W3, 64, 64, 3, 16);

  ProcW wEE = {eeW0, eeW1, eeW2, eeW3, Pf(EE_b0), Pf(EE_b1), Pf(EE_b2), Pf(EE_b3), Pf(EE_G), Pf(EE_B)};
  ProcW wNE = {neW0, neW1, neW2, neW3, Pf(NE_b0), Pf(NE_b1), Pf(NE_b2), Pf(NE_b3), Pf(NE_G), Pf(NE_B)};
  ProcW wEP = {epW0, epW1, epW2, epW3, Pf(EP_b0), Pf(EP_b1), Pf(EP_b2), Pf(EP_b3), Pf(EP_G), Pf(EP_B)};
  ProcW wNP = {npW0, npW1, npW2, npW3, Pf(NP_b0), Pf(NP_b1), Pf(NP_b2), Pf(NP_b3), Pf(NP_G), Pf(NP_B)};
  ProcW wND = {ndW0, ndW1, ndW2, ndW3, Pf(ND_b0), Pf(ND_b1), Pf(ND_b2), Pf(ND_b3), Pf(ND_b0), Pf(ND_b0)};

  const int eTiles = (E + 15) / 16, eBlocks = (eTiles + MLP_WAVES - 1) / MLP_WAVES;
  const int nTiles = (N + 15) / 16, nBlocks = (nTiles + MLP_WAVES - 1) / MLP_WAVES;
  const dim3 cvtGrid((N * 64 + 255) / 256);

  // encode
  k_edge_enc<<<eBlocks, 256, 0, stream>>>(nodesH, edgesH, senders, receivers,
                                          wEE, encE, agg0F, E);
  k_cvt_zero<<<cvtGrid, 256, 0, stream>>>(agg0F, agg0H, N * 64);
  k_node_enc<<<nBlocks, 256, 0, stream>>>(nodesH, agg0H, wNE, encN, N);

  // process (3 steps); step-0 latents alias the encoder outputs
  const _Float16* curN = encN;
  const _Float16* curE = encE;
  for (int s = 0; s < STEPS; ++s) {
    k_edge_proc<<<eBlocks, 256, 0, stream>>>(curN, curE, encN, encE, senders,
                                             receivers, wEP, latE, aggF, E);
    curE = latE;
    k_cvt_zero<<<cvtGrid, 256, 0, stream>>>(aggF, aggH, N * 64);
    k_node_proc<<<nBlocks, 256, 0, stream>>>(curN, aggH, encN, agg0H, wNP, latN, N);
    curN = latN;
  }

  // decode
  k_node_dec<<<nBlocks, 256, 0, stream>>>(curN, wND, out, N);
}
